// HalfBitLinear_77910706749690
// MI455X (gfx1250) — compile-verified, matrix operand order checked by reading, not run on hardware
//
#include <hip/hip_runtime.h>
#include <hip/hip_bf16.h>

typedef __bf16 bf16_t;
typedef __attribute__((ext_vector_type(16))) __bf16 v16bf;
typedef __attribute__((ext_vector_type(8)))  __bf16 v8bf;
typedef __attribute__((ext_vector_type(4)))  __bf16 v4bf;
typedef __attribute__((ext_vector_type(8)))  float  v8f;
typedef __attribute__((ext_vector_type(4)))  float  f32x4;

#define TOKENS 4096
#define IN_F   4096
#define OUT_F  16384

#define BM 128   // workgroup tile M
#define BN 256   // workgroup tile N
#define WM 64    // per-wave tile M
#define WN 64    // per-wave tile N

// ---------------------------------------------------------------- init
__global__ void hb_init_kernel(unsigned* __restrict__ maxbits) {
    if (threadIdx.x == 0 && blockIdx.x == 0) *maxbits = 0u;
}

// ---------------------------------------------------------------- max |w|
__global__ __launch_bounds__(256) void hb_maxabs_kernel(const float* __restrict__ w,
                                                        unsigned* __restrict__ maxbits,
                                                        long n4) {
    long tid    = (long)blockIdx.x * blockDim.x + threadIdx.x;
    long stride = (long)gridDim.x * blockDim.x;
    const f32x4* w4 = (const f32x4*)w;
    float m = 0.0f;
    for (long i = tid; i < n4; i += stride) {
        f32x4 v = __builtin_nontemporal_load(&w4[i]);   // streamed once: NT
        m = fmaxf(m, fmaxf(fmaxf(fabsf(v.x), fabsf(v.y)),
                           fmaxf(fabsf(v.z), fabsf(v.w))));
    }
    // wave32 reduction
    #pragma unroll
    for (int off = 16; off > 0; off >>= 1)
        m = fmaxf(m, __shfl_xor(m, off, 32));
    __shared__ float smax[8];
    int lane = threadIdx.x & 31;
    int wv   = threadIdx.x >> 5;
    if (lane == 0) smax[wv] = m;
    __syncthreads();
    if (threadIdx.x == 0) {
        float bm = smax[0];
        #pragma unroll
        for (int i = 1; i < 8; ++i) bm = fmaxf(bm, smax[i]);
        // |w| >= 0, so float bit-pattern order == unsigned order
        atomicMax(maxbits, __float_as_uint(bm));
    }
}

// ---------------------------------------------------------------- quantize w -> bf16 {0,1}
__global__ __launch_bounds__(256) void hb_quant_kernel(const float* __restrict__ w,
                                                       const float* __restrict__ noise,
                                                       const unsigned* __restrict__ maxbits,
                                                       bf16_t* __restrict__ qw,
                                                       long n4) {
    const float mx = __uint_as_float(*maxbits);
    long tid    = (long)blockIdx.x * blockDim.x + threadIdx.x;
    long stride = (long)gridDim.x * blockDim.x;
    const f32x4* w4 = (const f32x4*)w;
    const f32x4* r4 = (const f32x4*)noise;
    v4bf* q4 = (v4bf*)qw;
    const __bf16 one  = (__bf16)1.0f;
    const __bf16 zero = (__bf16)0.0f;
    for (long i = tid; i < n4; i += stride) {
        f32x4 wv = __builtin_nontemporal_load(&w4[i]);  // streamed once: NT
        f32x4 nv = __builtin_nontemporal_load(&r4[i]);  // streamed once: NT
        v4bf q;
        // keep = (w > 0) && (noise < |w|/max)  ==  (w > 0) && (noise*max < w)
        q.x = (wv.x > 0.0f && nv.x * mx < wv.x) ? one : zero;
        q.y = (wv.y > 0.0f && nv.y * mx < wv.y) ? one : zero;
        q.z = (wv.z > 0.0f && nv.z * mx < wv.z) ? one : zero;
        q.w = (wv.w > 0.0f && nv.w * mx < wv.w) ? one : zero;
        q4[i] = q;   // RT store: want qw resident in the 192MB L2 for the GEMM
    }
}

// ---------------------------------------------------------------- x f32 -> bf16
__global__ __launch_bounds__(256) void hb_cvtx_kernel(const float* __restrict__ x,
                                                      bf16_t* __restrict__ xb,
                                                      long n4) {
    long tid    = (long)blockIdx.x * blockDim.x + threadIdx.x;
    long stride = (long)gridDim.x * blockDim.x;
    const f32x4* x4 = (const f32x4*)x;
    v4bf* o4 = (v4bf*)xb;
    for (long i = tid; i < n4; i += stride) {
        f32x4 v = __builtin_nontemporal_load(&x4[i]);   // streamed once: NT
        v4bf o;
        o.x = (__bf16)v.x;
        o.y = (__bf16)v.y;
        o.z = (__bf16)v.z;
        o.w = (__bf16)v.w;
        o4[i] = o;   // RT store: want xb resident in L2 for the GEMM
    }
}

// ---------------------------------------------------------------- WMMA GEMM
// out[t,o] = sum_k xb[t,k] * qw[o,k] + bias[o]
__global__ __launch_bounds__(256) void hb_gemm_kernel(const bf16_t* __restrict__ xb,
                                                      const bf16_t* __restrict__ qw,
                                                      const float* __restrict__ bias,
                                                      float* __restrict__ out) {
    const int lane = threadIdx.x & 31;
    const int wave = threadIdx.x >> 5;
    const int wm   = wave >> 2;          // 0..1
    const int wn   = wave & 3;           // 0..3
    const int m0   = blockIdx.y * BM + wm * WM;
    const int n0   = blockIdx.x * BN + wn * WN;

    // 16-bit A/B fragment addressing (ISA 7.12.2):
    // lane l: row/col = l%16, K-chunks {kb..kb+7, kb+16..kb+23}, kb = (l/16)*8
    const int rc = lane & 15;
    const int kb = (lane >> 4) * 8;

    const bf16_t* arow[4];
    const bf16_t* brow[4];
    #pragma unroll
    for (int i = 0; i < 4; ++i) arow[i] = xb + (long)(m0 + i * 16 + rc) * IN_F;
    #pragma unroll
    for (int j = 0; j < 4; ++j) brow[j] = qw + (long)(n0 + j * 16 + rc) * IN_F;

    v8f acc[4][4] = {};

    #pragma unroll 2
    for (int kk = 0; kk < IN_F; kk += 32) {
        v16bf a[4], b[4];
        #pragma unroll
        for (int i = 0; i < 4; ++i) {
            v8bf lo = *(const v8bf*)(arow[i] + kk + kb);
            v8bf hi = *(const v8bf*)(arow[i] + kk + kb + 16);
            a[i] = __builtin_shufflevector(lo, hi, 0, 1, 2, 3, 4, 5, 6, 7,
                                                   8, 9, 10, 11, 12, 13, 14, 15);
        }
        #pragma unroll
        for (int j = 0; j < 4; ++j) {
            v8bf lo = *(const v8bf*)(brow[j] + kk + kb);
            v8bf hi = *(const v8bf*)(brow[j] + kk + kb + 16);
            b[j] = __builtin_shufflevector(lo, hi, 0, 1, 2, 3, 4, 5, 6, 7,
                                                   8, 9, 10, 11, 12, 13, 14, 15);
        }
        #pragma unroll
        for (int i = 0; i < 4; ++i)
            #pragma unroll
            for (int j = 0; j < 4; ++j)
                acc[i][j] = __builtin_amdgcn_wmma_f32_16x16x32_bf16(
                    false, a[i], false, b[j], (short)0, acc[i][j], false, false);
    }

    // Epilogue. C/D layout: elem e -> M = e + 8*(lane/16), N = lane%16
    // NT stores: out (256MB) is written once and never read; keep qw/xb in L2.
    const int row_half = (lane >> 4) * 8;
    const int col_in_t = lane & 15;
    #pragma unroll
    for (int j = 0; j < 4; ++j) {
        const int n = n0 + j * 16 + col_in_t;
        const float bj = bias[n];
        #pragma unroll
        for (int i = 0; i < 4; ++i) {
            const int mbase = m0 + i * 16 + row_half;
            #pragma unroll
            for (int e = 0; e < 8; ++e) {
                __builtin_nontemporal_store(acc[i][j][e] + bj,
                                            &out[(long)(mbase + e) * OUT_F + n]);
            }
        }
    }
}

// ---------------------------------------------------------------- launcher
extern "C" void kernel_launch(void* const* d_in, const int* in_sizes, int n_in,
                              void* d_out, int out_size, void* d_ws, size_t ws_size,
                              hipStream_t stream) {
    const float* x     = (const float*)d_in[0];   // [4096, 4096]
    const float* w     = (const float*)d_in[1];   // [16384, 4096]
    const float* bias  = (const float*)d_in[2];   // [16384]
    const float* noise = (const float*)d_in[3];   // [16384, 4096]
    float* out = (float*)d_out;                   // [4096, 16384]

    char* ws = (char*)d_ws;
    unsigned* maxbits = (unsigned*)ws;
    bf16_t* qw = (bf16_t*)(ws + 256);                                   // 128 MB
    bf16_t* xb = (bf16_t*)(ws + 256 + (size_t)OUT_F * IN_F * 2);        //  32 MB

    const long nW4 = (long)OUT_F * IN_F / 4;
    const long nX4 = (long)TOKENS * IN_F / 4;

    hb_init_kernel<<<1, 32, 0, stream>>>(maxbits);
    hb_maxabs_kernel<<<2048, 256, 0, stream>>>(w, maxbits, nW4);
    hb_quant_kernel<<<8192, 256, 0, stream>>>(w, noise, maxbits, qw, nW4);
    hb_cvtx_kernel<<<2048, 256, 0, stream>>>(x, xb, nX4);

    dim3 grid(OUT_F / BN, TOKENS / BM);   // (64, 32)
    hb_gemm_kernel<<<grid, 256, 0, stream>>>(xb, qw, bias, out);
}
